// customLSTM_40956808135044
// MI455X (gfx1250) — matrix operand order repaired and checked by reading, not compiled, not run on hardware
//
#include <hip/hip_runtime.h>
#include <hip/hip_bf16.h>

// ---------------------------------------------------------------------------
// LSTM cell, B=4096, DIN=1024, N=1024.
// z = [x|h0] @ [[W],[U]] + b  (one fused GEMM, K=2048, 4N=4096 columns)
// bf16 WMMA (v_wmma_f32_16x16x32_bf16) with f32 accumulate; all 4 gates are
// computed per block so the LSTM epilogue runs in-register.
// Round-2 changes: 2 row-fragments per wave (32x32 wave tile per gate) to
// halve LDS bytes per FLOP, and per-gate batched B-fragment loads so 4 WMMAs
// issue per s_wait_dscnt instead of 1.
// ---------------------------------------------------------------------------

#define Bb   4096
#define DIN  1024
#define NN   1024
#define KK   2048          // DIN + NN
#define FOURN 4096         // 4*NN
#define HN   4194304       // B*N  (output plane size)

typedef __attribute__((ext_vector_type(16))) __bf16 v16bf;
typedef __attribute__((ext_vector_type(8)))  float  v8f;

__device__ __forceinline__ unsigned short f2bf(float f) {
    unsigned u = __float_as_uint(f);
    unsigned r = u + 0x7FFFu + ((u >> 16) & 1u);   // round-to-nearest-even
    if ((u & 0x7F800000u) == 0x7F800000u) r = u;   // inf/nan: truncate
    return (unsigned short)(r >> 16);
}

__device__ __forceinline__ float hard_sigmoid(float z) {
    return fminf(fmaxf(0.2f * z + 0.5f, 0.0f), 1.0f);
}

// ---------------------------------------------------------------------------
// pack_A: A[m][k] = bf16( k<1024 ? x[m][k] : h0[m][k-1024] ), A is [4096,2048]
// ---------------------------------------------------------------------------
__global__ void lstm_pack_A(const float* __restrict__ x,
                            const float* __restrict__ h0,
                            unsigned short* __restrict__ A) {
    unsigned e = (blockIdx.x * blockDim.x + threadIdx.x) * 4u;   // flat elem idx
    unsigned m = e / KK;
    unsigned k = e % KK;
    const float* src = (k < DIN) ? (x + m * DIN + k) : (h0 + m * NN + (k - DIN));
    float4 v = *(const float4*)src;
    ushort4 o;
    o.x = f2bf(v.x); o.y = f2bf(v.y); o.z = f2bf(v.z); o.w = f2bf(v.w);
    *(ushort4*)(A + e) = o;
}

// ---------------------------------------------------------------------------
// pack_B: Ball[k][g*1024+n] = bf16( k<1024 ? Wg[k][n] : Ug[k-1024][n] )
// Ball is [2048, 4096] row-major, gate-major columns (f,i,o,c).
// ---------------------------------------------------------------------------
__global__ void lstm_pack_B(const float* __restrict__ Wf, const float* __restrict__ Wi,
                            const float* __restrict__ Wo, const float* __restrict__ Wc,
                            const float* __restrict__ Uf, const float* __restrict__ Ui,
                            const float* __restrict__ Uo, const float* __restrict__ Uc,
                            unsigned short* __restrict__ Ball) {
    unsigned e = (blockIdx.x * blockDim.x + threadIdx.x) * 4u;
    unsigned k = e / FOURN;
    unsigned c = e % FOURN;
    unsigned g = c / NN;
    unsigned n = c % NN;
    const float* W[4] = {Wf, Wi, Wo, Wc};
    const float* U[4] = {Uf, Ui, Uo, Uc};
    const float* src = (k < DIN) ? (W[g] + k * NN + n) : (U[g] + (k - DIN) * NN + n);
    float4 v = *(const float4*)src;
    ushort4 o;
    o.x = f2bf(v.x); o.y = f2bf(v.y); o.z = f2bf(v.z); o.w = f2bf(v.w);
    *(ushort4*)(Ball + e) = o;
}

// ---------------------------------------------------------------------------
// Fused GEMM + LSTM epilogue.
// Block: 256 threads (8 wave32).  Block tile = 256 rows x 32 cols, 4 gates.
// Wave tile = 32 rows x 32 cols per gate (2 row-frags x 2 col-frags).
// K-step = 32 (one WMMA K).  LDS stride 80 halves -> 32B-aligned fragments.
// ---------------------------------------------------------------------------
#define BMR  256    // block rows
#define BN   32     // block cols (per gate)
#define KS   32     // K per step (= WMMA K)
#define LDA  80     // padded LDS row stride in halves (160B)

__global__ void __launch_bounds__(256)
lstm_gemm_wmma(const unsigned short* __restrict__ A,     // [4096,2048] bf16
               const unsigned short* __restrict__ Ball,  // [2048,4096] bf16
               const float* __restrict__ c0,
               const float* __restrict__ bf_, const float* __restrict__ bi_,
               const float* __restrict__ bo_, const float* __restrict__ bc_,
               float* __restrict__ out) {
    __shared__ unsigned short As[BMR * LDA];         // 256 x 32 (padded) = 40KB
    __shared__ unsigned short Bs[4 * KS * LDA];      // 4 x 32 x 32 (padded) = 20KB

    const unsigned tid  = threadIdx.x;
    const unsigned lane = tid & 31u;
    const unsigned wave = tid >> 5u;                 // 0..7 -> 32-row slab
    const unsigned m0   = blockIdx.y * BMR;
    const unsigned n0   = blockIdx.x * BN;           // column within each gate

    // staging coords: A = one full 32-half row (64B) per thread
    //                 B = 16 halves (32B) per thread
    const unsigned bg   = tid >> 6u;                 // gate 0..3
    const unsigned bk   = (tid & 63u) >> 1u;         // 0..31
    const unsigned bcg  = tid & 1u;

    v8f acc[4][2][2];                                // [gate][rowfrag][colfrag]
#pragma unroll
    for (int g = 0; g < 4; ++g)
#pragma unroll
        for (int rf = 0; rf < 2; ++rf)
#pragma unroll
            for (int cf = 0; cf < 2; ++cf)
                acc[g][rf][cf] = (v8f){0.f, 0.f, 0.f, 0.f, 0.f, 0.f, 0.f, 0.f};

    for (unsigned k0 = 0; k0 < KK; k0 += KS) {
        // ---- stage A tile: 256 rows x 32 halves (64B per thread) ----
        {
            const uint4* src = (const uint4*)(A + (m0 + tid) * KK + k0);
            uint4* dst = (uint4*)(As + tid * LDA);
            dst[0] = src[0];
            dst[1] = src[1];
            dst[2] = src[2];
            dst[3] = src[3];
        }
        // ---- stage B tiles: 4 gates x 32x32 halves (32B per thread) ----
        {
            const uint4* src = (const uint4*)(Ball + (k0 + bk) * FOURN + bg * NN + n0 + bcg * 16u);
            uint4* dst = (uint4*)(Bs + (bg * KS + bk) * LDA + bcg * 16u);
            dst[0] = src[0];
            dst[1] = src[1];
        }
        __syncthreads();

        // A fragments: lane 0-15 -> M=lane,K=0..15 ; lane 16-31 -> M=lane-16,K=16..31
        const unsigned arow = wave * 32u + (lane & 15u);
        const unsigned akof = (lane >> 4u) * 16u;
        const v16bf a0 = *(const v16bf*)&As[(arow)       * LDA + akof];
        const v16bf a1 = *(const v16bf*)&As[(arow + 16u) * LDA + akof];

#pragma unroll
        for (int g = 0; g < 4; ++g) {
            // B fragments (lane -> N col, K = 16*(lane/16)+s), both col-halves
            v16bf b0, b1;
#pragma unroll
            for (int s = 0; s < 16; ++s) {
                const unsigned bro = (g * KS + (lane >> 4u) * 16u + s) * LDA + (lane & 15u);
                b0[s] = __builtin_bit_cast(__bf16, Bs[bro]);
                b1[s] = __builtin_bit_cast(__bf16, Bs[bro + 16u]);
            }
            acc[g][0][0] = __builtin_amdgcn_wmma_f32_16x16x32_bf16(
                false, a0, false, b0, (short)0, acc[g][0][0], false, false);
            acc[g][0][1] = __builtin_amdgcn_wmma_f32_16x16x32_bf16(
                false, a0, false, b1, (short)0, acc[g][0][1], false, false);
            acc[g][1][0] = __builtin_amdgcn_wmma_f32_16x16x32_bf16(
                false, a1, false, b0, (short)0, acc[g][1][0], false, false);
            acc[g][1][1] = __builtin_amdgcn_wmma_f32_16x16x32_bf16(
                false, a1, false, b1, (short)0, acc[g][1][1], false, false);
        }
        __syncthreads();
    }

    // ------------------- fused LSTM epilogue (in-register) -------------------
#pragma unroll
    for (int cf = 0; cf < 2; ++cf) {
        const unsigned col = n0 + cf * 16u + (lane & 15u);   // column within N
        const float vbf = bf_[col];
        const float vbi = bi_[col];
        const float vbo = bo_[col];
        const float vbc = bc_[col];
#pragma unroll
        for (int rf = 0; rf < 2; ++rf) {
            const unsigned rowBase = m0 + wave * 32u + rf * 16u + ((lane >> 4u) * 8u);
#pragma unroll
            for (int r = 0; r < 8; ++r) {
                const unsigned row = rowBase + r;
                const unsigned idx = row * NN + col;
                const float zf = acc[0][rf][cf][r] + vbf;
                const float zi = acc[1][rf][cf][r] + vbi;
                const float zo = acc[2][rf][cf][r] + vbo;
                const float zc = acc[3][rf][cf][r] + vbc;
                const float f  = hard_sigmoid(zf);
                const float i  = hard_sigmoid(zi);
                const float o  = hard_sigmoid(zo);
                const float ct = tanhf(zc);
                const float c  = f * c0[idx] + i * ct;
                const float h  = o * tanhf(c);
                out[idx]          = h;   // output = h
                out[HN + idx]     = c;   // new_c
                out[2 * HN + idx] = h;   // new_h
            }
        }
    }
}

// ---------------------------------------------------------------------------
// Launch
// ---------------------------------------------------------------------------
extern "C" void kernel_launch(void* const* d_in, const int* in_sizes, int n_in,
                              void* d_out, int out_size, void* d_ws, size_t ws_size,
                              hipStream_t stream) {
    const float* x  = (const float*)d_in[0];
    const float* c0 = (const float*)d_in[1];
    const float* h0 = (const float*)d_in[2];
    const float* Wf = (const float*)d_in[3];
    const float* Wi = (const float*)d_in[4];
    const float* Wo = (const float*)d_in[5];
    const float* Wc = (const float*)d_in[6];
    const float* Uf = (const float*)d_in[7];
    const float* Ui = (const float*)d_in[8];
    const float* Uo = (const float*)d_in[9];
    const float* Uc = (const float*)d_in[10];
    const float* bf = (const float*)d_in[11];
    const float* bi = (const float*)d_in[12];
    const float* bo = (const float*)d_in[13];
    const float* bc = (const float*)d_in[14];
    float* out = (float*)d_out;

    // Workspace layout: A bf16 [4096*2048] (16MB) | Ball bf16 [2048*4096] (16MB)
    unsigned short* Abf   = (unsigned short*)d_ws;
    unsigned short* Ballp = Abf + (size_t)Bb * KK;

    {   // pack A: 8M elems, 4 per thread
        const unsigned threads = (Bb * KK) / 4;
        lstm_pack_A<<<threads / 256, 256, 0, stream>>>(x, h0, Abf);
    }
    {   // pack B: 8M elems, 4 per thread
        const unsigned threads = (KK * FOURN) / 4;
        lstm_pack_B<<<threads / 256, 256, 0, stream>>>(Wf, Wi, Wo, Wc,
                                                       Uf, Ui, Uo, Uc, Ballp);
    }
    {   // fused GEMM + epilogue: grid = (N/BN, B/BMR) = (32, 16)
        dim3 grid(NN / BN, Bb / BMR);
        lstm_gemm_wmma<<<grid, 256, 0, stream>>>(Abf, Ballp, c0,
                                                 bf, bi, bo, bc, out);
    }
}
